// MultiHeadAttention_49263274885311
// MI455X (gfx1250) — matrix operand order repaired and checked by reading, not compiled, not run on hardware
//
#include <hip/hip_runtime.h>
#include <hip/hip_bf16.h>
#include <math.h>

// ---------------------------------------------------------------------------
// Causal MHA with KV cache, flash-attention style, CDNA5 (gfx1250) wave32 WMMA
// KC=64 keys/chunk: 16 WMMAs per chunk, ds_swizzle softmax reductions,
// masking only at the causal boundary, next-chunk global prefetch.
// ---------------------------------------------------------------------------

typedef __bf16 bf16;
typedef __attribute__((ext_vector_type(16))) __bf16 v16bf;
typedef __attribute__((ext_vector_type(8)))  __bf16 v8bf;
typedef __attribute__((ext_vector_type(8)))  float  v8f;
typedef __attribute__((ext_vector_type(2)))  __bf16 bf16x2;

constexpr int B  = 2;
constexpr int S  = 2048;
constexpr int D  = 1024;
constexpr int H  = 16;
constexpr int P  = 2048;
constexpr int DH = 64;      // head dim
constexpr int QB = 128;     // queries per block (8 waves x 16)
constexpr int QW = 16;      // queries per wave (one WMMA M-tile)
constexpr int KC = 64;      // keys per KV chunk

// reference multiplies scores by sqrt(dh)=8; fold 8*log2(e) into Q so the
// WMMA output is directly in the exp2 domain.
#define QSCALE 11.5415603271f   /* 8 * log2(e) */

template<int MASK>
__device__ __forceinline__ float swz_xor(float x) {
    // DS_SWIZZLE_B32 group-of-32: offset = {xor[14:10], or[9:5], and[4:0]}
    return __int_as_float(
        __builtin_amdgcn_ds_swizzle(__float_as_int(x), (MASK << 10) | 0x1F));
}
__device__ __forceinline__ float rowmax16(float x) {
    x = fmaxf(x, swz_xor<1>(x));
    x = fmaxf(x, swz_xor<2>(x));
    x = fmaxf(x, swz_xor<4>(x));
    x = fmaxf(x, swz_xor<8>(x));
    return x;
}
__device__ __forceinline__ float rowsum16(float x) {
    x += swz_xor<1>(x);
    x += swz_xor<2>(x);
    x += swz_xor<4>(x);
    x += swz_xor<8>(x);
    return x;
}

__global__ __launch_bounds__(256)
void attn_wmma_kernel(const float* __restrict__ x,
                      const float* __restrict__ past,
                      float* __restrict__ out)
{
    __shared__ bf16 kLds[KC * DH];        // K chunk, row-major  [key][dh]   8KB
    __shared__ bf16 vLds[DH * KC];        // V chunk, transposed [dh][key]   8KB
    __shared__ bf16 pLds[8 * QW * KC];    // per-wave P staging  [16][64]   16KB

    const int bh   = blockIdx.y;
    const int b    = bh / H;
    const int h    = bh % H;
    const int q0b  = blockIdx.x * QB;     // first query of the block
    const int tid  = threadIdx.x;
    const int wave = tid >> 5;
    const int lane = tid & 31;
    const int half = lane >> 4;           // 0 or 1
    const int l16  = lane & 15;
    const int q0   = q0b + wave * QW;     // first query of this wave's tile

    // ---- Q tile -> two A fragments (dh split into two 32-wide K slices) ----
    // Pre-scaled by 8*log2(e) so scores come out of WMMA in exp2 domain.
    v16bf aQ[2];
    {
        const float* qrow = x + ((size_t)(b * S + q0 + l16)) * (3 * D) + h * DH;
        #pragma unroll
        for (int hf = 0; hf < 2; ++hf) {
            #pragma unroll
            for (int v = 0; v < 8; ++v) {
                const int kp = ((v & 4) ? 16 : 0) + (half ? 8 : 0) + (v & 3) * 2;
                float2 f = *(const float2*)(qrow + hf * 32 + kp);
                aQ[hf][2 * v]     = (bf16)(f.x * QSCALE);
                aQ[hf][2 * v + 1] = (bf16)(f.y * QSCALE);
            }
        }
    }

    v8f  o[4] = {v8f{}, v8f{}, v8f{}, v8f{}};   // 16x64 f32 accumulator
    float rm[8], rl[8];                          // online-softmax state
    #pragma unroll
    for (int v = 0; v < 8; ++v) { rm[v] = -__builtin_inff(); rl[v] = 0.f; }

    const int kv_len = P + ((q0b + QB < S) ? (q0b + QB) : S);
    const int q_hi   = q0 + QW - 1;

    #pragma unroll 1
    for (int kv0 = 0; kv0 < kv_len; kv0 += KC) {
        __syncthreads();
        // ---- cooperative K/V chunk load: f32 -> bf16, V transposed --------
        {
            const int key = tid >> 2;             // 0..63
            const int cb  = (tid & 3) * 16;       // 0,16,32,48
            const int kg  = kv0 + key;
            const float *ksrc, *vsrc;
            if (kg < P) {
                ksrc = past + ((((size_t)b * 2 + 0) * H + h) * P + kg) * DH + cb;
                vsrc = past + ((((size_t)b * 2 + 1) * H + h) * P + kg) * DH + cb;
            } else {
                const int s = kg - P;
                const float* xr = x + ((size_t)(b * S + s)) * (3 * D) + h * DH + cb;
                ksrc = xr + D;
                vsrc = xr + 2 * D;
            }
            float4 k0 = *(const float4*)(ksrc);
            float4 k1 = *(const float4*)(ksrc + 4);
            float4 k2 = *(const float4*)(ksrc + 8);
            float4 k3 = *(const float4*)(ksrc + 12);
            float4 w0 = *(const float4*)(vsrc);
            float4 w1 = *(const float4*)(vsrc + 4);
            float4 w2 = *(const float4*)(vsrc + 8);
            float4 w3 = *(const float4*)(vsrc + 12);
            // K: packed 16B LDS stores (merge to ds_store_b128)
            bf16* kd = kLds + key * DH + cb;
            v8bf ka = {(bf16)k0.x, (bf16)k0.y, (bf16)k0.z, (bf16)k0.w,
                       (bf16)k1.x, (bf16)k1.y, (bf16)k1.z, (bf16)k1.w};
            v8bf kb = {(bf16)k2.x, (bf16)k2.y, (bf16)k2.z, (bf16)k2.w,
                       (bf16)k3.x, (bf16)k3.y, (bf16)k3.z, (bf16)k3.w};
            *(v8bf*)(kd)     = ka;
            *(v8bf*)(kd + 8) = kb;
            // V: transposed scatter
            const float wv[16] = {w0.x, w0.y, w0.z, w0.w, w1.x, w1.y, w1.z, w1.w,
                                  w2.x, w2.y, w2.z, w2.w, w3.x, w3.y, w3.z, w3.w};
            #pragma unroll
            for (int j = 0; j < 16; ++j)
                vLds[(cb + j) * KC + key] = (bf16)wv[j];
            // prefetch next chunk (speculative; dropped if invalid)
            const int kg2 = kg + KC;
            if (kg2 < P + S) {
                const float *kn, *vn;
                if (kg2 < P) {
                    kn = past + ((((size_t)b * 2 + 0) * H + h) * P + kg2) * DH + cb;
                    vn = past + ((((size_t)b * 2 + 1) * H + h) * P + kg2) * DH + cb;
                } else {
                    const int s2 = kg2 - P;
                    const float* xr2 = x + ((size_t)(b * S + s2)) * (3 * D) + h * DH + cb;
                    kn = xr2 + D;
                    vn = xr2 + 2 * D;
                }
                __builtin_prefetch(kn, 0, 0);
                __builtin_prefetch(vn, 0, 0);
            }
        }
        __syncthreads();

        if (kv0 > P + q_hi) continue;   // fully masked for this wave's tile

        // ---- QK^T: four 16-key groups, chained over the two dh halves -----
        v8f sc[4];
        #pragma unroll
        for (int g = 0; g < 4; ++g) {
            v16bf bK0, bK1;
            const int key = g * 16 + l16;
            #pragma unroll
            for (int v = 0; v < 8; ++v) {
                const int c0 = ((v & 4) ? 16 : 0) + (half ? 8 : 0) + (v & 3) * 2;
                bf16x2 p0 = *(const bf16x2*)(kLds + key * DH + c0);
                bf16x2 p1 = *(const bf16x2*)(kLds + key * DH + 32 + c0);
                bK0[2 * v] = p0.x; bK0[2 * v + 1] = p0.y;
                bK1[2 * v] = p1.x; bK1[2 * v + 1] = p1.y;
            }
            v8f z = {};
            z = __builtin_amdgcn_wmma_f32_16x16x32_bf16(false, aQ[0], false, bK0,
                                                        (short)0, z, false, false);
            z = __builtin_amdgcn_wmma_f32_16x16x32_bf16(false, aQ[1], false, bK1,
                                                        (short)0, z, false, false);
            sc[g] = z;
        }

        // ---- causal mask: only chunks touching the boundary need it -------
        if (kv0 + (KC - 1) > P + q0) {
            #pragma unroll
            for (int v = 0; v < 8; ++v) {
                const int lim = P + q0 + v + 8 * half;   // last visible key
                #pragma unroll
                for (int g = 0; g < 4; ++g)
                    if (kv0 + g * 16 + l16 > lim) sc[g][v] = -1e9f;
            }
        }

        // ---- online softmax (exp2 domain; scale folded into Q) ------------
        bf16* pr = pLds + wave * (QW * KC);
        #pragma unroll
        for (int v = 0; v < 8; ++v) {
            const float mx = rowmax16(fmaxf(fmaxf(sc[0][v], sc[1][v]),
                                            fmaxf(sc[2][v], sc[3][v])));
            const float nm = fmaxf(rm[v], mx);
            const float alpha = exp2f(rm[v] - nm);
            const float p0 = exp2f(sc[0][v] - nm);
            const float p1 = exp2f(sc[1][v] - nm);
            const float p2 = exp2f(sc[2][v] - nm);
            const float p3 = exp2f(sc[3][v] - nm);
            const float rs = rowsum16((p0 + p1) + (p2 + p3));
            rm[v] = nm;
            rl[v] = rl[v] * alpha + rs;
            o[0][v] *= alpha; o[1][v] *= alpha; o[2][v] *= alpha; o[3][v] *= alpha;
            bf16* row = pr + (v + 8 * half) * KC;
            row[l16]      = (bf16)p0;
            row[16 + l16] = (bf16)p1;
            row[32 + l16] = (bf16)p2;
            row[48 + l16] = (bf16)p3;
        }

        // ---- read P back in A layout (two 32-K fragments), then P x V -----
        v16bf aP[2];
        #pragma unroll
        for (int f = 0; f < 2; ++f)
            #pragma unroll
            for (int v = 0; v < 8; ++v) {
                const int kp = f * 32 + ((v & 4) ? 16 : 0) + (half ? 8 : 0) + (v & 3) * 2;
                bf16x2 pp = *(const bf16x2*)(pr + l16 * KC + kp);
                aP[f][2 * v] = pp.x; aP[f][2 * v + 1] = pp.y;
            }
        #pragma unroll
        for (int t = 0; t < 4; ++t) {
            const int nd = t * 16 + l16;
            v16bf bV0, bV1;
            #pragma unroll
            for (int v = 0; v < 8; ++v) {
                const int kk = ((v & 4) ? 16 : 0) + (half ? 8 : 0) + (v & 3) * 2;
                bf16x2 a0 = *(const bf16x2*)(vLds + nd * KC + kk);
                bf16x2 a1 = *(const bf16x2*)(vLds + nd * KC + 32 + kk);
                bV0[2 * v] = a0.x; bV0[2 * v + 1] = a0.y;
                bV1[2 * v] = a1.x; bV1[2 * v + 1] = a1.y;
            }
            o[t] = __builtin_amdgcn_wmma_f32_16x16x32_bf16(false, aP[0], false, bV0,
                                                           (short)0, o[t], false, false);
            o[t] = __builtin_amdgcn_wmma_f32_16x16x32_bf16(false, aP[1], false, bV1,
                                                           (short)0, o[t], false, false);
        }
    }

    // ---- normalize and store out[b, q, h*DH + col] -------------------------
    #pragma unroll
    for (int t = 0; t < 4; ++t) {
        #pragma unroll
        for (int v = 0; v < 8; ++v) {
            const int q   = q0 + v + 8 * half;
            const int col = h * DH + t * 16 + l16;
            out[((size_t)(b * S + q)) * D + col] = o[t][v] / rl[v];
        }
    }
}

// present[b, t, h, s, c] = x[b, s, (1+t)*D + h*DH + c]   (float4 vectorized)
__global__ __launch_bounds__(256)
void present_kernel(const float* __restrict__ x, float* __restrict__ present)
{
    const size_t total4 = (size_t)B * 2 * H * S * (DH / 4);
    size_t i = (size_t)blockIdx.x * blockDim.x + threadIdx.x;
    if (i >= total4) return;
    const int c4 = (int)(i % (DH / 4));
    const int s  = (int)((i / (DH / 4)) % S);
    const int h  = (int)((i / ((size_t)(DH / 4) * S)) % H);
    const int t  = (int)((i / ((size_t)(DH / 4) * S * H)) % 2);
    const int b  = (int)( i / ((size_t)(DH / 4) * S * H * 2));
    const float4 val = *(const float4*)(x + ((size_t)(b * S + s)) * (3 * D)
                                          + (1 + t) * D + h * DH + c4 * 4);
    *(float4*)(present + i * 4) = val;
}

extern "C" void kernel_launch(void* const* d_in, const int* in_sizes, int n_in,
                              void* d_out, int out_size, void* d_ws, size_t ws_size,
                              hipStream_t stream)
{
    const float* x    = (const float*)d_in[0];
    const float* past = (const float*)d_in[1];
    // d_in[2] (mask) is recomputed analytically in-kernel.
    float* out     = (float*)d_out;
    float* present = out + (size_t)B * S * D;

    dim3 grid(S / QB, B * H);           // 16 x 32 blocks, 8 waves each
    attn_wmma_kernel<<<grid, 256, 0, stream>>>(x, past, out);

    const size_t total4 = (size_t)B * 2 * H * S * (DH / 4);
    present_kernel<<<(unsigned)((total4 + 255) / 256), 256, 0, stream>>>(x, present);
}
